// GNNModel_31825707663693
// MI455X (gfx1250) — matrix operand order repaired and checked by reading, not compiled, not run on hardware
//
#include <hip/hip_runtime.h>

// ---------------------------------------------------------------------------
// GCN (2x GCNConv + global_add_pool + MLP head) for MI455X / gfx1250.
//   N=100000 nodes, E=1000000 edges, F=128, H=64, G=128 graphs, C=1.
// Strategy:
//   * bf16 WMMA (v_wmma_f32_16x16x32_bf16) for all dense GEMMs, f32 accum.
//   * Edge scatter via f32 global atomics: both operand arrays are 25.6MB,
//     fully resident in the 192MB L2 where the atomic units live.
//   * 64 lanes per edge -> coalesced 256B row gather/update.
// ---------------------------------------------------------------------------

#define N_NODES 100000
#define N_EDGES 1000000
#define F_IN    128
#define H_DIM   64
#define G_BATCH 128

typedef __attribute__((ext_vector_type(16))) __bf16 bf16x16;
typedef __attribute__((ext_vector_type(8)))  __bf16 bf16x8;
typedef __attribute__((ext_vector_type(8)))  float  f32x8;

union ABFrag { bf16x16 v; bf16x8 h[2]; };

// ---------------- elementwise / graph kernels ------------------------------

__global__ void k_fill(float* __restrict__ p, float v, int n) {
  int i = blockIdx.x * blockDim.x + threadIdx.x;
  if (i < n) p[i] = v;
}

__global__ void k_deg_count(float* __restrict__ deg, const int* __restrict__ dst, int e) {
  int i = blockIdx.x * blockDim.x + threadIdx.x;
  if (i < e) atomicAdd(&deg[dst[i]], 1.0f);
}

__global__ void k_rsqrt_inplace(float* __restrict__ p, int n) {
  int i = blockIdx.x * blockDim.x + threadIdx.x;
  if (i < n) p[i] = rsqrtf(p[i]);
}

__global__ void k_cvt_bf16(const float* __restrict__ s, __bf16* __restrict__ d, int n) {
  int i = blockIdx.x * blockDim.x + threadIdx.x;
  if (i < n) d[i] = (__bf16)s[i];
}

// agg[i,f] = h[i,f] * dis[i]^2   (self-loop term initializes the accumulator)
__global__ void k_self_init(const float* __restrict__ hf, const float* __restrict__ dis,
                            float* __restrict__ agg) {
  int i = blockIdx.x * blockDim.x + threadIdx.x;   // exactly N*H threads
  int node = i >> 6;
  float ds = dis[node];
  agg[i] = hf[i] * ds * ds;
}

// agg[dst,f] += h[src,f] * dis[src]*dis[dst]; 64 lanes per edge (f = lane%64)
__global__ void k_scatter(const float* __restrict__ hf, float* __restrict__ agg,
                          const float* __restrict__ dis, const int* __restrict__ src,
                          const int* __restrict__ dst) {
  int t = blockIdx.x * blockDim.x + threadIdx.x;   // exactly E*64 threads
  int e = t >> 6;
  int f = t & 63;
  int s = src[e];
  int d = dst[e];
  int e2 = e + 16;
  if (e2 < N_EDGES)  // pull a future source row toward L0/L2 (global_prefetch_b8)
    __builtin_prefetch(&hf[(size_t)src[e2] * 64 + f], 0, 0);
  float norm = dis[s] * dis[d];
  atomicAdd(&agg[(size_t)d * 64 + f], hf[(size_t)s * 64 + f] * norm);
}

// hb = bf16(relu(agg + b))  — fused epilogue + precision cast for next GEMM
__global__ void k_relu_bias_cvt(const float* __restrict__ agg, const float* __restrict__ bias,
                                __bf16* __restrict__ hb) {
  int i = blockIdx.x * blockDim.x + threadIdx.x;
  int f = i & 63;
  float v = agg[i] + bias[f];
  hb[i] = (__bf16)fmaxf(v, 0.0f);
}

// g[batch[node],f] += relu(agg + b)  (batch is sorted -> near-sequential targets)
__global__ void k_relu_bias_pool(const float* __restrict__ agg, const float* __restrict__ bias,
                                 const int* __restrict__ batch, float* __restrict__ g) {
  int i = blockIdx.x * blockDim.x + threadIdx.x;
  int node = i >> 6;
  int f = i & 63;
  float v = fmaxf(agg[i] + bias[f], 0.0f);
  atomicAdd(&g[(size_t)batch[node] * 64 + f], v);
}

// out[i] = sum_j relu(gh[i,j] + bl1[j]) * Wl2[j] + bl2
__global__ void k_final(const float* __restrict__ gh, const float* __restrict__ bl1,
                        const float* __restrict__ Wl2, const float* __restrict__ bl2,
                        float* __restrict__ out) {
  int i = threadIdx.x;   // 128 threads
  float acc = 0.0f;
  for (int j = 0; j < 64; ++j) {
    float v = fmaxf(gh[i * 64 + j] + bl1[j], 0.0f);
    acc += v * Wl2[j];
  }
  out[i] = acc + bl2[0];
}

// ---------------- WMMA bf16 GEMM: C[M,64] = A[M,K](bf16) x W[K,64](f32) -----
// 8 waves/block, each wave one 16x16 f32 tile; block covers 32 rows x 64 cols.
// W is staged once per block into LDS in the exact CDNA5 B-fragment layout
// (bf16-converted), so per K-step each lane does two aligned 16B LDS reads.
__global__ __launch_bounds__(256) void k_gemm_bf16(
    const __bf16* __restrict__ A,   // [M, K] row-major bf16
    const float*  __restrict__ W,   // [K, 64] row-major f32
    float*        __restrict__ C,   // [M, 64] row-major f32
    int K) {                        // K in {64, 128}, multiple of 32
  __shared__ __align__(16) __bf16 lb[8192];  // 4 col-tiles * (K/32) ksteps * 32 lanes * 16
  const int ksteps = K >> 5;
  const int tid = threadIdx.x;

  // Stage W -> LDS in B-fragment order: element (nt, ks, lane, j) holds
// W[ks*32 + (j<8 ? 0 : 16) + (lane<16 ? 0 : 8) + j%8, nt*16 + lane%16]  (ISA 16-bit B 32x16 layout).
  for (int idx = tid; idx < K * 64; idx += 256) {
    int j    = idx & 15;
    int lane = (idx >> 4) & 31;
    int t2   = idx >> 9;
    int ks   = t2 % ksteps;
    int nt   = t2 / ksteps;
    int k = ks * 32 + ((j & 8) ? 16 : 0) + ((lane & 16) ? 8 : 0) + (j & 7);
    int n = nt * 16 + (lane & 15);
    lb[idx] = (__bf16)W[k * 64 + n];
  }
  __syncthreads();

  const int wave = tid >> 5;
  const int lane = tid & 31;
  const int wm   = wave >> 2;                 // 0..1 row tile
  const int wn   = wave & 3;                  // 0..3 col tile
  const int row0 = blockIdx.x * 32 + wm * 16;
  const int arow = row0 + (lane & 15);
  const int khalf = (lane & 16) ? 8 : 0;      // ISA 16-bit A 16x32 lane split

  f32x8 acc = {};
  const __bf16* arowp = A + (size_t)arow * K;
  for (int ks = 0; ks < ksteps; ++ks) {
    ABFrag a, b;
    const __bf16* ap = arowp + ks * 32 + khalf;          // K = khalf..khalf+7
    a.h[0] = *reinterpret_cast<const bf16x8*>(ap);
    a.h[1] = *reinterpret_cast<const bf16x8*>(ap + 16);  // K = 16+khalf..+7
    const __bf16* bp = &lb[(((wn * ksteps + ks) * 32) + lane) * 16];
    b.h[0] = *reinterpret_cast<const bf16x8*>(bp);
    b.h[1] = *reinterpret_cast<const bf16x8*>(bp + 8);
    acc = __builtin_amdgcn_wmma_f32_16x16x32_bf16(false, a.v, false, b.v,
                                                  (short)0, acc, false, false);
  }

  // 8-VGPR f32 C/D layout: VGPR r -> row (r or r+8 by lane half), col = lane%16
  const int col   = wn * 16 + (lane & 15);
  const int rbase = row0 + ((lane & 16) ? 8 : 0);
  float* cp = C + (size_t)rbase * 64 + col;
#pragma unroll
  for (int r = 0; r < 8; ++r) cp[(size_t)r * 64] = acc[r];
}

// ---------------------------------------------------------------------------

extern "C" void kernel_launch(void* const* d_in, const int* in_sizes, int n_in,
                              void* d_out, int out_size, void* d_ws, size_t ws_size,
                              hipStream_t stream) {
  (void)in_sizes; (void)n_in; (void)out_size; (void)ws_size;
  const float* x     = (const float*)d_in[0];
  const int*   ei    = (const int*)d_in[1];
  const int*   batch = (const int*)d_in[2];
  const float* W1    = (const float*)d_in[3];
  const float* b1    = (const float*)d_in[4];
  const float* W2    = (const float*)d_in[5];
  const float* b2    = (const float*)d_in[6];
  const float* Wl1   = (const float*)d_in[7];
  const float* bl1   = (const float*)d_in[8];
  const float* Wl2   = (const float*)d_in[9];
  const float* bl2   = (const float*)d_in[10];
  const int* srcI = ei;
  const int* dstI = ei + N_EDGES;
  float* out = (float*)d_out;

  // workspace carve-out (256B aligned slabs, ~90MB total)
  char* ws = (char*)d_ws;
  size_t o = 0;
  auto alloc = [&](size_t bytes) {
    void* p = ws + o;
    o = (o + bytes + 255) & ~(size_t)255;
    return p;
  };
  float*  dis = (float*) alloc((size_t)N_NODES * 4);           // deg -> rsqrt(deg)
  __bf16* xb  = (__bf16*)alloc((size_t)N_NODES * F_IN * 2);    // bf16 features
  __bf16* hb  = (__bf16*)alloc((size_t)N_NODES * H_DIM * 2);   // bf16 activations
  float*  hf  = (float*) alloc((size_t)N_NODES * H_DIM * 4);   // GEMM output (f32)
  float*  agg = (float*) alloc((size_t)N_NODES * H_DIM * 4);   // scatter accumulator
  float*  g   = (float*) alloc((size_t)G_BATCH * H_DIM * 4);   // pooled graphs
  __bf16* gb  = (__bf16*)alloc((size_t)G_BATCH * H_DIM * 2);
  float*  gh  = (float*) alloc((size_t)G_BATCH * H_DIM * 4);

  const int B = 256;
  const int nhB = (N_NODES * H_DIM) / B;       // 25000, exact
  const int efB = (int)(((size_t)N_EDGES * 64) / B);  // 250000, exact

  // ---- degrees: deg = 1 + in-degree(dst); dis = rsqrt(deg) ----
  k_fill<<<(N_NODES + B - 1) / B, B, 0, stream>>>(dis, 1.0f, N_NODES);
  k_deg_count<<<(N_EDGES + B - 1) / B, B, 0, stream>>>(dis, dstI, N_EDGES);
  k_rsqrt_inplace<<<(N_NODES + B - 1) / B, B, 0, stream>>>(dis, N_NODES);

  // ---- x -> bf16 (halves the dominant HBM stream) ----
  k_cvt_bf16<<<(N_NODES * F_IN) / B, B, 0, stream>>>(x, xb, N_NODES * F_IN);

  // ---- layer 1: h = relu( scatter(X@W1) + self + b1 ) ----
  k_gemm_bf16<<<N_NODES / 32, B, 0, stream>>>(xb, W1, hf, F_IN);
  k_self_init<<<nhB, B, 0, stream>>>(hf, dis, agg);
  k_scatter<<<efB, B, 0, stream>>>(hf, agg, dis, srcI, dstI);
  k_relu_bias_cvt<<<nhB, B, 0, stream>>>(agg, b1, hb);

  // ---- layer 2 ----
  k_gemm_bf16<<<N_NODES / 32, B, 0, stream>>>(hb, W2, hf, H_DIM);
  k_self_init<<<nhB, B, 0, stream>>>(hf, dis, agg);
  k_scatter<<<efB, B, 0, stream>>>(hf, agg, dis, srcI, dstI);

  // ---- global_add_pool (fused with relu+bias of layer 2) ----
  k_fill<<<(G_BATCH * H_DIM + B - 1) / B, B, 0, stream>>>(g, 0.0f, G_BATCH * H_DIM);
  k_relu_bias_pool<<<nhB, B, 0, stream>>>(agg, b2, batch, g);

  // ---- MLP head: relu(g@Wl1+bl1) @ Wl2 + bl2 ----
  k_cvt_bf16<<<(G_BATCH * H_DIM + B - 1) / B, B, 0, stream>>>(g, gb, G_BATCH * H_DIM);
  k_gemm_bf16<<<G_BATCH / 32, B, 0, stream>>>(gb, Wl1, gh, H_DIM);
  k_final<<<1, G_BATCH, 0, stream>>>(gh, bl1, Wl2, bl2, out);
}
